// transformer_mu_55791625175536
// MI455X (gfx1250) — compile-verified, matrix operand order checked by reading, not run on hardware
//
#include <hip/hip_runtime.h>
#include <math.h>

// ---------------------------------------------------------------------------
// Point-transformer layer for MI455X (gfx1250, wave32, WMMA 16x16x32 f16).
// v2: all weights pre-transposed+converted to f16 once (B-frags = 2x b128),
//     all intermediates stored f16 (halves HBM traffic; op is memory-bound:
//     ~250MB @ 23.3TB/s ~ 11us vs ~20 GFLOP of WMMA work which is trivial),
//     KNN stages positions into LDS via async global->LDS loads (ASYNCcnt).
// Pipeline:
//   0. prep_wt_kernel    : f32 weights -> f16 transposed (w_start/k/q/v, pw2 pad, aw1, aw2)
//   1. knn_kernel        : async-LDS staged brute-force top-16 KNN
//   2. gemm_start_kernel : xs = x @ w_start + b            (WMMA, f16 out)
//   3. gemm_kqv_kernel   : key/qry/val = xs @ w_* + b      (WMMA, A reused 3x)
//   4. pos_mlp1_kernel   : pe1_pre = pos_rel @ pw1 + pb1   (f16 out, padded 21->32)
//   5. bn_stats_kernel   : per-channel mean/var -> fused scale/shift
//   6. pe_gemm_kernel    : pe = relu(bn(pe1_pre)) @ pw2 + pb2   (WMMA)
//   7. attn1_gemm_kernel : h1 = (q - k[idx] + pe) @ aw1 + ab1   (WMMA, packed-f16 A build)
//   8. bn_stats_kernel   : BN stats for h1
//   9. attn2_kernel      : softmax((relu(bn(h1))@aw2+ab2)/8); out = sum a*(v+pe) + x
// Workspace: ~280 MB.
// ---------------------------------------------------------------------------

typedef __attribute__((ext_vector_type(16))) _Float16 v16h;
typedef __attribute__((ext_vector_type(8)))  _Float16 v8h;
typedef __attribute__((ext_vector_type(8)))  float    v8f;

#define WMMA_F16(a, b, c) \
  __builtin_amdgcn_wmma_f32_16x16x32_f16(false, (a), false, (b), (short)0, (c), false, false)

constexpr int B_  = 4;
constexpr int N_  = 4096;
constexpr int K_  = 16;
constexpr int CIN = 192;
constexpr int DIM = 192;
constexpr int H_  = 3;
constexpr int D_  = 64;    // per-head dim
constexpr int PH  = 21;    // pos hidden
constexpr int PHP = 32;    // pos hidden padded row stride (16B-aligned f16 rows)
constexpr int AH  = 64;    // attn hidden
constexpr int M_  = B_ * N_;        // 16384 dense rows
constexpr int R_  = B_ * N_ * K_;   // 262144 neighbor rows
constexpr float EPS_ = 1e-3f;

// ---------------- WMMA fragment helpers (CDNA5 ISA 7.12.2 layouts) ----

// Row-major f16 fragment load (used for A from activations and B from
// pre-transposed weights Wt[n][k]): lane&15 = row/col, two b128 loads.
__device__ __forceinline__ v16h load_rm_f16(const _Float16* __restrict__ p, int ld,
                                            int row0, int k0, int lane) {
  const int half = (lane >> 4) & 1;
  const _Float16* q = p + (size_t)(row0 + (lane & 15)) * ld + k0 + half * 8;
  v8h lo = *(const v8h*)q;          // K = k0+half*8 + {0..7}
  v8h hi = *(const v8h*)(q + 16);   // K = k0+half*8 + {16..23}
  return __builtin_shufflevector(lo, hi, 0, 1, 2, 3, 4, 5, 6, 7,
                                 8, 9, 10, 11, 12, 13, 14, 15);
}

// A-frag from row-major f32 (only for the raw input x).
__device__ __forceinline__ v16h load_a_f32(const float* __restrict__ p, int ld,
                                           int row0, int k0, int lane) {
  const int half = (lane >> 4) & 1;
  const float* q = p + (size_t)(row0 + (lane & 15)) * ld + k0 + half * 8;
  v16h a;
#pragma unroll
  for (int i = 0; i < 8; ++i) a[i] = (_Float16)q[i];
#pragma unroll
  for (int i = 0; i < 8; ++i) a[8 + i] = (_Float16)q[16 + i];
  return a;
}

// C/D store + bias -> f16: VGPR i -> row m0 + half*8 + i, col n0 + (lane&15).
__device__ __forceinline__ void store_c_bias_f16(_Float16* __restrict__ out, int ldo,
                                                 int m0, int n0, int lane, v8f c,
                                                 const float* __restrict__ bias) {
  const int half = (lane >> 4) & 1;
  const int n = n0 + (lane & 15);
  const float bv = bias[n];
#pragma unroll
  for (int i = 0; i < 8; ++i)
    out[(size_t)(m0 + half * 8 + i) * ldo + n] = (_Float16)(c[i] + bv);
}

// ---------------- 0. weight prep: f32 [bh][Kd][Nd] -> f16 [bh][Nd][Kpad] --
__global__ void prep_wt_kernel(const float* __restrict__ w, _Float16* __restrict__ wt,
                               int Kd, int Nd, int Kpad, int batch) {
  const int tid = blockIdx.x * 256 + threadIdx.x;
  const int per = Nd * Kpad;
  if (tid >= batch * per) return;
  const int bh = tid / per, r = tid % per;
  const int n = r / Kpad, k = r % Kpad;
  const float v = (k < Kd) ? w[(size_t)bh * Kd * Nd + (size_t)k * Nd + n] : 0.f;
  wt[(size_t)bh * per + r] = (_Float16)v;
}

// ---------------- 1. KNN: async global->LDS staging, register top-16 ------
__global__ void knn_kernel(const float* __restrict__ pos, int* __restrict__ idx) {
  __shared__ float sp[N_ * 3];  // 48 KB of one batch's positions
  const int b = blockIdx.x / (N_ / 256);
  const int chunk = blockIdx.x % (N_ / 256);
  const float* pb = pos + (size_t)b * N_ * 3;
  // Stage via CDNA5 async loads (tracked by ASYNCcnt, no VGPR round-trip).
  for (int c = threadIdx.x; c < (N_ * 3) / 4; c += 256) {
    unsigned lds = (unsigned)(uintptr_t)(&sp[4 * c]);   // low 32 bits = LDS offset
    unsigned long long ga = (unsigned long long)(uintptr_t)(pb + 4 * c);
    asm volatile("global_load_async_to_lds_b128 %0, %1, off"
                 :: "v"(lds), "v"(ga) : "memory");
  }
  asm volatile("s_wait_asynccnt 0" ::: "memory");
  __syncthreads();
  const int n = chunk * 256 + threadIdx.x;
  const float qx = sp[n * 3], qy = sp[n * 3 + 1], qz = sp[n * 3 + 2];
  float bd[K_]; int bi[K_];
#pragma unroll
  for (int j = 0; j < K_; ++j) { bd[j] = 3.4e38f; bi[j] = 0; }
  for (int m = 0; m < N_; ++m) {
    float dx = sp[m * 3] - qx, dy = sp[m * 3 + 1] - qy, dz = sp[m * 3 + 2] - qz;
    float d2 = dx * dx + dy * dy + dz * dz;
    if (d2 < bd[K_ - 1]) {           // register-only sorted insertion chain
      float v = d2; int vi = m;
#pragma unroll
      for (int j = 0; j < K_; ++j) {
        if (v < bd[j]) { float t = bd[j]; int ti = bi[j]; bd[j] = v; bi[j] = vi; v = t; vi = ti; }
      }
    }
  }
  int* op = idx + ((size_t)b * N_ + n) * K_;
#pragma unroll
  for (int j = 0; j < K_; ++j) op[j] = bi[j];
}

// ---------------- 2. xs = x @ w_start + b (f32 in, f16 out) ---------------
__global__ void gemm_start_kernel(const float* __restrict__ x, const _Float16* __restrict__ wt,
                                  const float* __restrict__ bias, _Float16* __restrict__ xs) {
  const int lane = threadIdx.x & 31;
  const int wid = blockIdx.x * 8 + (threadIdx.x >> 5);
  const int m0 = (wid / (DIM / 16)) * 16;
  const int n0 = (wid % (DIM / 16)) * 16;
  v8f c = {};
#pragma unroll
  for (int kt = 0; kt < CIN / 32; ++kt) {
    __builtin_prefetch(x + (size_t)(m0 + (lane & 15)) * CIN + kt * 32 + 64, 0, 0);
    v16h a = load_a_f32(x, CIN, m0, kt * 32, lane);
    v16h bm = load_rm_f16(wt, CIN, n0, kt * 32, lane);
    c = WMMA_F16(a, bm, c);
  }
  store_c_bias_f16(xs, DIM, m0, n0, lane, c, bias);
}

// ---------------- 3. key/qry/val = xs @ w_{k,q,v} + b ---------------------
__global__ void gemm_kqv_kernel(const _Float16* __restrict__ xs,
                                const _Float16* __restrict__ wtk, const float* __restrict__ bk,
                                const _Float16* __restrict__ wtq, const float* __restrict__ bq,
                                const _Float16* __restrict__ wtv, const float* __restrict__ bv,
                                _Float16* __restrict__ keyo, _Float16* __restrict__ qryo,
                                _Float16* __restrict__ valo) {
  const int lane = threadIdx.x & 31;
  const int wid = blockIdx.x * 8 + (threadIdx.x >> 5);
  const int m0 = (wid / (DIM / 16)) * 16;
  const int n0 = (wid % (DIM / 16)) * 16;
  v8f ck = {}, cq = {}, cv = {};
#pragma unroll
  for (int kt = 0; kt < DIM / 32; ++kt) {
    v16h a = load_rm_f16(xs, DIM, m0, kt * 32, lane);   // A reused across 3 GEMMs
    ck = WMMA_F16(a, load_rm_f16(wtk, DIM, n0, kt * 32, lane), ck);
    cq = WMMA_F16(a, load_rm_f16(wtq, DIM, n0, kt * 32, lane), cq);
    cv = WMMA_F16(a, load_rm_f16(wtv, DIM, n0, kt * 32, lane), cv);
  }
  store_c_bias_f16(keyo, DIM, m0, n0, lane, ck, bk);
  store_c_bias_f16(qryo, DIM, m0, n0, lane, cq, bq);
  store_c_bias_f16(valo, DIM, m0, n0, lane, cv, bv);
}

// ---------------- 4. pe1_pre = pos_rel @ pw1 + pb1 (21ch, padded to 32) ---
__global__ void pos_mlp1_kernel(const float* __restrict__ pos, const int* __restrict__ idx,
                                const float* __restrict__ pw1, const float* __restrict__ pb1,
                                _Float16* __restrict__ pe1) {
  const int tid = blockIdx.x * 256 + threadIdx.x;   // H_*R_ threads
  const int h = tid / R_;
  const int r = tid % R_;
  const int p = r >> 4;          // flattened (b,n)
  const int b = p / N_;
  const int nb = idx[r];
  const size_t prow = (size_t)b * N_ + nb;
  const float dx = pos[(size_t)p * 3 + 0] - pos[prow * 3 + 0];
  const float dy = pos[(size_t)p * 3 + 1] - pos[prow * 3 + 1];
  const float dz = pos[(size_t)p * 3 + 2] - pos[prow * 3 + 2];
  const float* w = pw1 + h * 3 * PH;
  const float* bb = pb1 + h * PH;
  _Float16* o = pe1 + ((size_t)h * R_ + r) * PHP;
#pragma unroll
  for (int ch = 0; ch < PH; ++ch)
    o[ch] = (_Float16)(dx * w[ch] + dy * w[PH + ch] + dz * w[2 * PH + ch] + bb[ch]);
#pragma unroll
  for (int ch = PH; ch < PHP; ++ch) o[ch] = (_Float16)0.f;   // deterministic pad
}

// ---------------- 5/8. BN stats: one block per (head,channel) -------------
// src layout [H][R_][S] f16 (C valid channels); fused scale/shift output.
__global__ void bn_stats_kernel(const _Float16* __restrict__ src, int C, int S,
                                const float* __restrict__ gamma,
                                const float* __restrict__ beta,
                                float* __restrict__ scale, float* __restrict__ shift) {
  const int hc = blockIdx.x;
  const int h = hc / C, ch = hc % C;
  const _Float16* base = src + (size_t)h * R_ * S + ch;
  float s = 0.f, s2 = 0.f;
  for (int r = threadIdx.x; r < R_; r += 256) {
    float v = (float)base[(size_t)r * S];
    s += v; s2 += v * v;
  }
  __shared__ float ss[256], ss2[256];
  ss[threadIdx.x] = s; ss2[threadIdx.x] = s2;
  __syncthreads();
  for (int o = 128; o > 0; o >>= 1) {
    if (threadIdx.x < o) {
      ss[threadIdx.x] += ss[threadIdx.x + o];
      ss2[threadIdx.x] += ss2[threadIdx.x + o];
    }
    __syncthreads();
  }
  if (threadIdx.x == 0) {
    float mean = ss[0] * (1.f / (float)R_);
    float var  = ss2[0] * (1.f / (float)R_) - mean * mean;
    float sc = gamma[hc] * rsqrtf(var + EPS_);
    scale[hc] = sc;
    shift[hc] = beta[hc] - mean * sc;
  }
}

// ---------------- 6. pe = relu(bn(pe1_pre)) @ pw2 + pb2 -------------------
__global__ void pe_gemm_kernel(const _Float16* __restrict__ pe1, const float* __restrict__ scale,
                               const float* __restrict__ shift, const _Float16* __restrict__ wt_pw2,
                               const float* __restrict__ pb2, _Float16* __restrict__ pe) {
  const int lane = threadIdx.x & 31;
  const int wid = blockIdx.x * 8 + (threadIdx.x >> 5);
  const int tilesPerHead = (R_ / 16) * (D_ / 16);
  const int h = wid / tilesPerHead;
  const int rem = wid % tilesPerHead;
  const int m0 = (rem / (D_ / 16)) * 16;
  const int n0 = (rem % (D_ / 16)) * 16;
  const int half = (lane >> 4) & 1;
  const _Float16* row = pe1 + ((size_t)h * R_ + m0 + (lane & 15)) * PHP;
  const float* sc = scale + h * PH;
  const float* sh = shift + h * PH;
  v16h a;
#pragma unroll
  for (int i = 0; i < 16; ++i) {          // BN + ReLU fused into A-frag build
    int k = half * 8 + ((i < 8) ? 0 : 16) + (i & 7);
    float v = 0.f;
    if (k < PH) {                          // pad channels must stay exactly 0
      v = fmaf((float)row[k], sc[k], sh[k]);
      v = v > 0.f ? v : 0.f;
    }
    a[i] = (_Float16)v;
  }
  v16h bm = load_rm_f16(wt_pw2 + (size_t)h * D_ * PHP, PHP, n0, 0, lane);  // padded K
  v8f c = {};
  c = WMMA_F16(a, bm, c);
  store_c_bias_f16(pe + (size_t)h * R_ * D_, D_, m0, n0, lane, c, pb2 + h * D_);
}

// ---------------- 7. h1 = (q - k[idx] + pe) @ aw1 + ab1 -------------------
__global__ void attn1_gemm_kernel(const _Float16* __restrict__ qry, const _Float16* __restrict__ key,
                                  const int* __restrict__ idx, const _Float16* __restrict__ pe,
                                  const _Float16* __restrict__ wt_aw1, const float* __restrict__ ab1,
                                  _Float16* __restrict__ h1) {
  const int lane = threadIdx.x & 31;
  const int wid = blockIdx.x * 8 + (threadIdx.x >> 5);
  const int tilesPerHead = (R_ / 16) * (AH / 16);
  const int h = wid / tilesPerHead;
  const int rem = wid % tilesPerHead;
  const int m0 = (rem / (AH / 16)) * 16;
  const int n0 = (rem % (AH / 16)) * 16;
  const int half = (lane >> 4) & 1;
  const int r = m0 + (lane & 15);
  const int p = r >> 4;
  const int b = p / N_;
  const int nb = idx[r];
  const _Float16* qrow = qry + (size_t)p * DIM + h * D_;
  const _Float16* krow = key + ((size_t)b * N_ + nb) * DIM + h * D_;
  const _Float16* prow = pe + ((size_t)h * R_ + r) * D_;
  const _Float16* w = wt_aw1 + (size_t)h * D_ * AH;
  v8f c = {};
#pragma unroll
  for (int kt = 0; kt < D_ / 32; ++kt) {
    const int k0 = kt * 32 + half * 8;
    // q - gather(k) + pe built with packed-f16 vector math (v_pk_add_f16)
    v8h alo = *(const v8h*)(qrow + k0) - *(const v8h*)(krow + k0) + *(const v8h*)(prow + k0);
    v8h ahi = *(const v8h*)(qrow + k0 + 16) - *(const v8h*)(krow + k0 + 16) +
              *(const v8h*)(prow + k0 + 16);
    v16h a = __builtin_shufflevector(alo, ahi, 0, 1, 2, 3, 4, 5, 6, 7,
                                     8, 9, 10, 11, 12, 13, 14, 15);
    c = WMMA_F16(a, load_rm_f16(w, AH, n0, kt * 32, lane), c);
  }
  store_c_bias_f16(h1 + (size_t)h * R_ * AH, AH, m0, n0, lane, c, ab1 + h * AH);
}

// ------- 9. a2 = (relu(bn(h1)) @ aw2 + ab2)/8; softmax(ch); aggregate -----
// One wave = one (head, point): its 16x64 WMMA output is the full attention
// matrix (16 neighbors x 64 channels) -> softmax + neighbor-sum in registers.
__global__ void attn2_kernel(const _Float16* __restrict__ h1, const float* __restrict__ scale,
                             const float* __restrict__ shift, const _Float16* __restrict__ wt_aw2,
                             const float* __restrict__ ab2, const _Float16* __restrict__ val,
                             const _Float16* __restrict__ pe, const float* __restrict__ x,
                             float* __restrict__ out) {
  const int lane = threadIdx.x & 31;
  const int wid = blockIdx.x * 8 + (threadIdx.x >> 5);
  const int h = wid / (R_ / 16);
  const int p = wid % (R_ / 16);          // flattened (b,n); 16 rows = all K_=16 neighbors
  const int m0 = p * 16;
  const int half = (lane >> 4) & 1;
  const int l = lane & 15;
  const _Float16* srow = h1 + ((size_t)h * R_ + m0 + l) * AH;
  const float* sc = scale + h * AH;
  const float* sh = shift + h * AH;
  const _Float16* w = wt_aw2 + (size_t)h * AH * D_;
  v8f c0 = {}, c1 = {}, c2 = {}, c3 = {};
#pragma unroll
  for (int kt = 0; kt < AH / 32; ++kt) {
    v16h a;
#pragma unroll
    for (int i = 0; i < 16; ++i) {        // BN + ReLU fused into A-frag build
      int k = kt * 32 + half * 8 + ((i < 8) ? 0 : 16) + (i & 7);
      float v = fmaf((float)srow[k], sc[k], sh[k]);
      a[i] = (_Float16)(v > 0.f ? v : 0.f);
    }
    c0 = WMMA_F16(a, load_rm_f16(w, AH, 0,  kt * 32, lane), c0);
    c1 = WMMA_F16(a, load_rm_f16(w, AH, 16, kt * 32, lane), c1);
    c2 = WMMA_F16(a, load_rm_f16(w, AH, 32, kt * 32, lane), c2);
    c3 = WMMA_F16(a, load_rm_f16(w, AH, 48, kt * 32, lane), c3);
  }
  const float* bb = ab2 + h * D_;
  float aj[4][8];
  const float b0 = bb[l], b1 = bb[16 + l], b2 = bb[32 + l], b3 = bb[48 + l];
#pragma unroll
  for (int i = 0; i < 8; ++i) {
    aj[0][i] = (c0[i] + b0) * 0.125f;     // 1/sqrt(D_=64)
    aj[1][i] = (c1[i] + b1) * 0.125f;
    aj[2][i] = (c2[i] + b2) * 0.125f;
    aj[3][i] = (c3[i] + b3) * 0.125f;
  }
  // Channel softmax per row: reduce across the 16 lanes of each half-wave.
#pragma unroll
  for (int i = 0; i < 8; ++i) {
    float mx = fmaxf(fmaxf(aj[0][i], aj[1][i]), fmaxf(aj[2][i], aj[3][i]));
#pragma unroll
    for (int o = 1; o < 16; o <<= 1) mx = fmaxf(mx, __shfl_xor(mx, o, 32));
    float e0 = __expf(aj[0][i] - mx), e1 = __expf(aj[1][i] - mx);
    float e2 = __expf(aj[2][i] - mx), e3 = __expf(aj[3][i] - mx);
    float s = e0 + e1 + e2 + e3;
#pragma unroll
    for (int o = 1; o < 16; o <<= 1) s += __shfl_xor(s, o, 32);
    float inv = 1.f / s;
    aj[0][i] = e0 * inv; aj[1][i] = e1 * inv; aj[2][i] = e2 * inv; aj[3][i] = e3 * inv;
  }
  // Aggregate over neighbors: out[d] = sum_r a[r][d] * (val[d] + pe[r][d])
  const _Float16* vrow = val + (size_t)p * DIM + h * D_;
  const _Float16* pbase = pe + ((size_t)h * R_ + m0 + half * 8) * D_;
  float acc[4];
#pragma unroll
  for (int j = 0; j < 4; ++j) {
    const int d = j * 16 + l;
    const float vv = (float)vrow[d];
    float s = 0.f;
#pragma unroll
    for (int i = 0; i < 8; ++i)
      s += aj[j][i] * (vv + (float)pbase[(size_t)i * D_ + d]);
    acc[j] = s;
  }
#pragma unroll
  for (int j = 0; j < 4; ++j) acc[j] += __shfl_xor(acc[j], 16, 32);  // rows 0-7 + 8-15
  if (lane < 16) {
    const float* xrow = x + (size_t)p * CIN + h * D_;
    float* orow = out + (size_t)p * DIM + h * D_;
#pragma unroll
    for (int j = 0; j < 4; ++j)
      orow[j * 16 + l] = acc[j] + xrow[j * 16 + l];   // + residual x
  }
}

// ---------------------------------------------------------------------------
extern "C" void kernel_launch(void* const* d_in, const int* in_sizes, int n_in,
                              void* d_out, int out_size, void* d_ws, size_t ws_size,
                              hipStream_t stream) {
  const float* x       = (const float*)d_in[0];
  const float* pos     = (const float*)d_in[1];
  const float* w_start = (const float*)d_in[2];
  const float* b_start = (const float*)d_in[3];
  const float* w_key   = (const float*)d_in[4];
  const float* b_key   = (const float*)d_in[5];
  const float* w_qry   = (const float*)d_in[6];
  const float* b_qry   = (const float*)d_in[7];
  const float* w_val   = (const float*)d_in[8];
  const float* b_val   = (const float*)d_in[9];
  const float* pw1     = (const float*)d_in[10];
  const float* pb1     = (const float*)d_in[11];
  const float* pg1     = (const float*)d_in[12];
  const float* pbe1    = (const float*)d_in[13];
  const float* pw2     = (const float*)d_in[14];
  const float* pb2     = (const float*)d_in[15];
  const float* aw1     = (const float*)d_in[16];
  const float* ab1     = (const float*)d_in[17];
  const float* ag1     = (const float*)d_in[18];
  const float* abe1    = (const float*)d_in[19];
  const float* aw2     = (const float*)d_in[20];
  const float* ab2     = (const float*)d_in[21];
  float* out = (float*)d_out;

  char* ws = (char*)d_ws;
  size_t off = 0;
  auto alloc = [&](size_t bytes) -> void* {
    off = (off + 255) & ~(size_t)255;
    void* pp = ws + off;
    off += bytes;
    return pp;
  };
  typedef _Float16 f16;
  int* idx       = (int*)alloc((size_t)R_ * sizeof(int));
  f16* wt_start  = (f16*)alloc((size_t)CIN * DIM * sizeof(f16));
  f16* wt_key    = (f16*)alloc((size_t)DIM * DIM * sizeof(f16));
  f16* wt_qry    = (f16*)alloc((size_t)DIM * DIM * sizeof(f16));
  f16* wt_val    = (f16*)alloc((size_t)DIM * DIM * sizeof(f16));
  f16* wt_pw2    = (f16*)alloc((size_t)H_ * D_ * PHP * sizeof(f16));
  f16* wt_aw1    = (f16*)alloc((size_t)H_ * AH * D_ * sizeof(f16));
  f16* wt_aw2    = (f16*)alloc((size_t)H_ * D_ * AH * sizeof(f16));
  f16* xs        = (f16*)alloc((size_t)M_ * DIM * sizeof(f16));
  f16* keyb      = (f16*)alloc((size_t)M_ * DIM * sizeof(f16));
  f16* qryb      = (f16*)alloc((size_t)M_ * DIM * sizeof(f16));
  f16* valb      = (f16*)alloc((size_t)M_ * DIM * sizeof(f16));
  f16* pe1       = (f16*)alloc((size_t)H_ * R_ * PHP * sizeof(f16));
  f16* peb       = (f16*)alloc((size_t)H_ * R_ * D_ * sizeof(f16));
  f16* h1b       = (f16*)alloc((size_t)H_ * R_ * AH * sizeof(f16));
  float* bnp_scale = (float*)alloc((size_t)H_ * PH * sizeof(float));
  float* bnp_shift = (float*)alloc((size_t)H_ * PH * sizeof(float));
  float* bna_scale = (float*)alloc((size_t)H_ * AH * sizeof(float));
  float* bna_shift = (float*)alloc((size_t)H_ * AH * sizeof(float));
  (void)ws_size; (void)in_sizes; (void)n_in; (void)out_size;

  // 0. one-time weight transpose + f16 convert (tiny)
  prep_wt_kernel<<<(CIN * DIM + 255) / 256, 256, 0, stream>>>(w_start, wt_start, CIN, DIM, CIN, 1);
  prep_wt_kernel<<<(DIM * DIM + 255) / 256, 256, 0, stream>>>(w_key, wt_key, DIM, DIM, DIM, 1);
  prep_wt_kernel<<<(DIM * DIM + 255) / 256, 256, 0, stream>>>(w_qry, wt_qry, DIM, DIM, DIM, 1);
  prep_wt_kernel<<<(DIM * DIM + 255) / 256, 256, 0, stream>>>(w_val, wt_val, DIM, DIM, DIM, 1);
  prep_wt_kernel<<<(H_ * D_ * PHP + 255) / 256, 256, 0, stream>>>(pw2, wt_pw2, PH, D_, PHP, H_);
  prep_wt_kernel<<<(H_ * AH * D_ + 255) / 256, 256, 0, stream>>>(aw1, wt_aw1, D_, AH, D_, H_);
  prep_wt_kernel<<<(H_ * D_ * AH + 255) / 256, 256, 0, stream>>>(aw2, wt_aw2, AH, D_, AH, H_);

  // 1. KNN (async LDS staging)
  knn_kernel<<<B_ * (N_ / 256), 256, 0, stream>>>(pos, idx);

  // 2. xs = x @ w_start + b   (1024 x 12 tiles, 8 waves/block)
  gemm_start_kernel<<<(M_ / 16) * (DIM / 16) / 8, 256, 0, stream>>>(x, wt_start, b_start, xs);

  // 3. key/qry/val
  gemm_kqv_kernel<<<(M_ / 16) * (DIM / 16) / 8, 256, 0, stream>>>(
      xs, wt_key, b_key, wt_qry, b_qry, wt_val, b_val, keyb, qryb, valb);

  // 4. pos-MLP layer 1 pre-BN (padded f16 rows)
  pos_mlp1_kernel<<<(H_ * R_) / 256, 256, 0, stream>>>(pos, idx, pw1, pb1, pe1);

  // 5. BN stats for pos-MLP (3*21 channels, row stride 32)
  bn_stats_kernel<<<H_ * PH, 256, 0, stream>>>(pe1, PH, PHP, pg1, pbe1, bnp_scale, bnp_shift);

  // 6. pe = relu(bn(pe1)) @ pw2 + pb2
  pe_gemm_kernel<<<H_ * (R_ / 16) * (D_ / 16) / 8, 256, 0, stream>>>(
      pe1, bnp_scale, bnp_shift, wt_pw2, pb2, peb);

  // 7. h1 = (q - k[idx] + pe) @ aw1 + ab1
  attn1_gemm_kernel<<<H_ * (R_ / 16) * (AH / 16) / 8, 256, 0, stream>>>(
      qryb, keyb, idx, peb, wt_aw1, ab1, h1b);

  // 8. BN stats for attention MLP (3*64 channels)
  bn_stats_kernel<<<H_ * AH, 256, 0, stream>>>(h1b, AH, AH, ag1, abe1, bna_scale, bna_shift);

  // 9. second attn matmul + softmax + neighbor aggregation + residual
  attn2_kernel<<<H_ * (R_ / 16) / 8, 256, 0, stream>>>(
      h1b, bna_scale, bna_shift, wt_aw2, ab2, valb, peb, x, out);
}
